// PhysNetCore_29626684408242
// MI455X (gfx1250) — matrix operand order repaired and checked by reading, not compiled, not run on hardware
//
#include <hip/hip_runtime.h>
#include <math.h>

#define F_DIM   128
#define K_RBF   64
#define LOG2F_  0.69314718056f
#define LOG2E_  1.44269504089f
#define CUTOFF_ 10.0f
#define EXPM10  4.5399929762484854e-05f   // exp(-10)

typedef __attribute__((ext_vector_type(16))) _Float16 v16h;
typedef __attribute__((ext_vector_type(8)))  _Float16 h8;
typedef __attribute__((ext_vector_type(2)))  __fp16   fp16x2;   // cvt_pkrtz return type
typedef __attribute__((ext_vector_type(8)))  float    v8f;

// shifted softplus: log(1+exp(x)) - log(2).
// Raw-HW form: max(x,0) + ln2*(log2(1 + exp2(-|x|*log2e)) - 1).
// Argument of log2 is in (1,2] -> no denormal fixup needed, so use the raw
// v_exp_f32 / v_log_f32 builtins (2 TRANS + ~4 VALU, no cmp/cndmask).
__device__ __forceinline__ float ssp(float x) {
    float e = __builtin_amdgcn_exp2f(-fabsf(x) * LOG2E_);
    float l = __builtin_amdgcn_logf(1.0f + e);          // log2(1+e), arg in (1,2]
    return fmaxf(x, 0.0f) + LOG2F_ * (l - 1.0f);
}

// ---------------------------------------------------------------------------
// X[a,f] = embedding_table[Za[a], f]
// ---------------------------------------------------------------------------
__global__ void embed_kernel(const float* __restrict__ emb, const int* __restrict__ Za,
                             float* __restrict__ X, int n) {
    int i = blockIdx.x * blockDim.x + threadIdx.x;
    if (i < n * F_DIM) {
        int a = i >> 7, f = i & 127;
        X[i] = emb[Za[a] * F_DIM + f];
    }
}

// ---------------------------------------------------------------------------
// o = a + b (elementwise, float4; n4 = count/4)
// ---------------------------------------------------------------------------
__global__ void add4_kernel(const float4* __restrict__ a, const float4* __restrict__ b,
                            float4* __restrict__ o, int n4) {
    int i = blockIdx.x * blockDim.x + threadIdx.x;
    if (i < n4) {
        float4 x = a[i], y = b[i];
        float4 r = {x.x + y.x, x.y + y.y, x.z + y.z, x.w + y.w};
        o[i] = r;
    }
}

// ---------------------------------------------------------------------------
// Build one A fragment (16 rows x 32 K) for v_wmma_f32_16x16x32_f16.
// ISA A layout: lane_lo = M row; elements 0..7 -> K = k0..k0+7,
// elements 8..15 -> K = k0+16..k0+23, where k0 = chunk*32 + lane_hi*8.
// f32->f16 via v_cvt_pk_rtz_f16_f32 (no half-register move packing).
// ---------------------------------------------------------------------------
template<int PRE>
__device__ __forceinline__ v16h load_afrag(const float* __restrict__ src, int k0) {
    float4 u0 = *(const float4*)(src + k0);
    float4 u1 = *(const float4*)(src + k0 + 4);
    float4 u2 = *(const float4*)(src + k0 + 16);
    float4 u3 = *(const float4*)(src + k0 + 20);
    float t[16] = {u0.x,u0.y,u0.z,u0.w, u1.x,u1.y,u1.z,u1.w,
                   u2.x,u2.y,u2.z,u2.w, u3.x,u3.y,u3.z,u3.w};
    if (PRE) {
#pragma unroll
        for (int e = 0; e < 16; e++) t[e] = ssp(t[e]);
    }
    union { v16h v; fp16x2 p[8]; } r;
#pragma unroll
    for (int j = 0; j < 8; j++)
        r.p[j] = __builtin_amdgcn_cvt_pkrtz(t[2*j], t[2*j + 1]);
    return r.v;
}

// ---------------------------------------------------------------------------
// n-tile sweep (8 tiles x 4 K-chunks -> 8 WMMAs per tile) with the bounds
// handling resolved at compile time (FULL = no guards at all).
// ---------------------------------------------------------------------------
template<int POST, int RES, bool FULL>
__device__ __forceinline__ void gemm_tiles(
        const h8* __restrict__ wp, const float* __restrict__ sb,
        const float* __restrict__ su, const v16h* afrag0, const v16h* afrag1,
        const float* __restrict__ resv, float* __restrict__ out,
        int nrows, int m_base, int lane, int lane_lo, int lane_hi) {
    for (int nt = 0; nt < 8; nt++) {
        int ncol = nt * 16 + lane_lo;
        v8f acc0 = {}, acc1 = {};
#pragma unroll
        for (int c = 0; c < 4; c++) {
            union { v16h v; h8 h[2]; } b;
            int base = ((nt * 4 + c) * 32 + lane) * 2;   // in h8 units
            b.h[0] = wp[base];
            b.h[1] = wp[base + 1];
            acc0 = __builtin_amdgcn_wmma_f32_16x16x32_f16(
                       false, afrag0[c], false, b.v, (short)0, acc0, false, false);
            acc1 = __builtin_amdgcn_wmma_f32_16x16x32_f16(
                       false, afrag1[c], false, b.v, (short)0, acc1, false, false);
        }
        const float bv = sb[ncol];
        const float uv = (RES == 2) ? su[ncol] : 0.0f;
        const int r0base = m_base + lane_hi * 8;

        // D layout: row = 8*lane_hi + v (strip 0) / +16 (strip 1), col = lane_lo
#pragma unroll
        for (int v = 0; v < 8; v++) {
            int mr0 = r0base + v;
            int mr1 = mr0 + 16;
            if (FULL || mr0 < nrows) {
                float o = acc0[v] + bv;
                if (POST) o = ssp(o);
                long oi = (long)mr0 * F_DIM + ncol;
                if (RES == 1)      o += resv[oi];
                else if (RES == 2) o += uv * resv[oi];
                out[oi] = o;
            }
            if (FULL || mr1 < nrows) {
                float o = acc1[v] + bv;
                if (POST) o = ssp(o);
                long oi = (long)mr1 * F_DIM + ncol;
                if (RES == 1)      o += resv[oi];
                else if (RES == 2) o += uv * resv[oi];
                out[oi] = o;
            }
        }
    }
}

// ---------------------------------------------------------------------------
// Generic 128-wide dense layer via WMMA f16->f32 (compile-time variants):
//   out = [POST]( [PRE](in) @ W + bias )  (+ res if RES==1 | + u*res if RES==2)
// Block: 256 threads = 8 waves; each wave owns a 32-row strip (two 16-row
// A-fragment sets).  W is pre-packed in LDS into the exact B-fragment register
// image so each B fragment is two ds_load_b128 (32 contiguous bytes per lane).
// ---------------------------------------------------------------------------
template<int PRE, int POST, int RES>
__global__ void __launch_bounds__(256)
gemm128_wmma(const float* __restrict__ in, const float* __restrict__ W,
             const float* __restrict__ bias, const float* __restrict__ resv,
             const float* __restrict__ uvec, float* __restrict__ out,
             int nrows) {
    __shared__ _Float16 sraw[F_DIM * F_DIM];              // raw W, f16, row-major [k][n]
    __shared__ __align__(16) _Float16 sWp[F_DIM * F_DIM]; // packed B-fragment image
    __shared__ float sb[F_DIM];
    __shared__ float su[F_DIM];

    // stage 1: coalesced f32->f16 copy of W
    for (int i = threadIdx.x; i < F_DIM * F_DIM; i += 256)
        sraw[i] = (_Float16)W[i];
    if (threadIdx.x < F_DIM) {
        sb[threadIdx.x] = bias[threadIdx.x];
        if (RES == 2) su[threadIdx.x] = uvec[threadIdx.x];
    }
    __syncthreads();

    // stage 2: pack into B-fragment layout.
    // p = (((nt*4 + c)*32 + lane)*16 + t)
    // fragment element t of (nt,c,lane): k = c*32 + (lane>>4)*8 + (t&7) + (t>>3)*16, n = nt*16 + (lane&15)
    for (int p = threadIdx.x; p < F_DIM * F_DIM; p += 256) {
        int t    = p & 15;
        int ln   = (p >> 4) & 31;
        int c    = (p >> 9) & 3;
        int nt   = p >> 11;
        int k    = c * 32 + (ln >> 4) * 8 + (t & 7) + (t >> 3) * 16;
        int n    = nt * 16 + (ln & 15);
        sWp[p] = sraw[k * F_DIM + n];
    }
    __syncthreads();

    const int wave    = threadIdx.x >> 5;
    const int lane    = threadIdx.x & 31;
    const int lane_lo = lane & 15;
    const int lane_hi = lane >> 4;
    const int m_base  = blockIdx.x * 256 + wave * 32;   // 32 rows per wave
    if (m_base >= nrows) return;   // wave-uniform exit: EXEC stays full below

    // A fragments for the two 16-row strips (rows clamped; OOB masked at store)
    int arow0 = m_base + lane_lo;       if (arow0 > nrows - 1) arow0 = nrows - 1;
    int arow1 = m_base + 16 + lane_lo;  if (arow1 > nrows - 1) arow1 = nrows - 1;
    const float* src0 = in + (long)arow0 * F_DIM;
    const float* src1 = in + (long)arow1 * F_DIM;

    v16h afrag0[4], afrag1[4];
#pragma unroll
    for (int c = 0; c < 4; c++) {
        int k0 = c * 32 + lane_hi * 8;
        afrag0[c] = load_afrag<PRE>(src0, k0);
        afrag1[c] = load_afrag<PRE>(src1, k0);
    }

    const h8* wp = (const h8*)sWp;   // 16-byte granules

    if ((m_base + 32) <= nrows) {    // wave-uniform: fast path, no store guards
        gemm_tiles<POST, RES, true >(wp, sb, su, afrag0, afrag1, resv, out,
                                     nrows, m_base, lane, lane_lo, lane_hi);
    } else {                         // boundary block only
        gemm_tiles<POST, RES, false>(wp, sb, su, afrag0, afrag1, resv, out,
                                     nrows, m_base, lane, lane_lo, lane_hi);
    }
}

// ---------------------------------------------------------------------------
// Fused edge kernel: distance -> cutoff -> 64 RBFs -> g = rbf @ Wg ->
// atomicAdd(MSG[idx_i], g * XJ[idx_j]).  One wave per edge; rbf values are
// exchanged between lanes with __shfl; Wg (64x128 f32, 32KB) staged in LDS.
// ---------------------------------------------------------------------------
__global__ void __launch_bounds__(256)
msg_kernel(const float* __restrict__ Ra, const int* __restrict__ idx_i,
           const int* __restrict__ idx_j, const float* __restrict__ Wg,
           const float* __restrict__ XJ, float* __restrict__ MSG, int E) {
    __shared__ __align__(16) float sWg[K_RBF * F_DIM];
    for (int i = threadIdx.x; i < K_RBF * F_DIM; i += 256) sWg[i] = Wg[i];
    __syncthreads();
    const float4* wg4 = (const float4*)sWg;   // [64][32] float4

    const int lane = threadIdx.x & 31;
    int gw = (blockIdx.x * blockDim.x + threadIdx.x) >> 5;
    int nw = (gridDim.x * blockDim.x) >> 5;

    const float cstep = (EXPM10 - 1.0f) / 63.0f;          // linspace(1, e^-10, 64) step
    float wtmp = 0.5f / ((1.0f - EXPM10) / (float)K_RBF);
    const float width = wtmp * wtmp;
    const float c0 = 1.0f + (float)lane * cstep;
    const float c1 = 1.0f + (float)(lane + 32) * cstep;

    for (int e = gw; e < E; e += nw) {
        int ai = idx_i[e], aj = idx_j[e];
        float dx = Ra[aj * 3 + 0] - Ra[ai * 3 + 0];
        float dy = Ra[aj * 3 + 1] - Ra[ai * 3 + 1];
        float dz = Ra[aj * 3 + 2] - Ra[ai * 3 + 2];
        float D  = sqrtf(dx * dx + dy * dy + dz * dz + 1e-12f);
        float xr = D * (1.0f / CUTOFF_);
        float xr2 = xr * xr, xr3 = xr2 * xr;
        float fcut = (D < CUTOFF_)
                   ? 1.0f - 6.0f * xr3 * xr2 + 15.0f * xr2 * xr2 - 10.0f * xr3 : 0.0f;
        float eD = __builtin_amdgcn_exp2f(-D * LOG2E_);
        float d0 = eD - c0, d1 = eD - c1;
        float r0 = __builtin_amdgcn_exp2f(-width * d0 * d0 * LOG2E_) * fcut;  // rbf[lane]
        float r1 = __builtin_amdgcn_exp2f(-width * d1 * d1 * LOG2E_) * fcut;  // rbf[lane+32]

        float4 acc = {0.f, 0.f, 0.f, 0.f};            // g for cols lane*4 .. lane*4+3
#pragma unroll 8
        for (int k = 0; k < 32; k++) {
            float bk = __shfl(r0, k, 32);
            float4 w = wg4[k * 32 + lane];
            acc.x += bk * w.x; acc.y += bk * w.y; acc.z += bk * w.z; acc.w += bk * w.w;
        }
#pragma unroll 8
        for (int k = 0; k < 32; k++) {
            float bk = __shfl(r1, k, 32);
            float4 w = wg4[(k + 32) * 32 + lane];
            acc.x += bk * w.x; acc.y += bk * w.y; acc.z += bk * w.z; acc.w += bk * w.w;
        }
        float4 xjv = ((const float4*)(XJ + (long)aj * F_DIM))[lane];
        float* dst = MSG + (long)ai * F_DIM + lane * 4;
        atomicAdd(dst + 0, acc.x * xjv.x);
        atomicAdd(dst + 1, acc.y * xjv.y);
        atomicAdd(dst + 2, acc.z * xjv.z);
        atomicAdd(dst + 3, acc.w * xjv.w);
    }
}

// ---------------------------------------------------------------------------
// Output head: out = act(h) @ Wout[128,2]; Ea/Qa accumulation + nhloss term.
// One wave per atom; shuffle reduction.
// ---------------------------------------------------------------------------
__global__ void __launch_bounds__(256)
out_kernel(const float* __restrict__ H, const float* __restrict__ Wout,
           float* __restrict__ EaQa, float* __restrict__ nh,
           const float* __restrict__ last2, float* __restrict__ cur2,
           int n, int first) {
    int w = (blockIdx.x * blockDim.x + threadIdx.x) >> 5;
    int lane = threadIdx.x & 31;
    if (w >= n) return;
    const float* hr = H + (long)w * F_DIM;
    float4 hv = ((const float4*)hr)[lane];
    float a0 = ssp(hv.x), a1 = ssp(hv.y), a2 = ssp(hv.z), a3 = ssp(hv.w);
    // Wout row-major [128][2] -> float4 covers two rows
    float4 w01 = ((const float4*)Wout)[lane * 2 + 0];
    float4 w23 = ((const float4*)Wout)[lane * 2 + 1];
    float s0 = a0 * w01.x + a1 * w01.z + a2 * w23.x + a3 * w23.z;
    float s1 = a0 * w01.y + a1 * w01.w + a2 * w23.y + a3 * w23.w;
#pragma unroll
    for (int off = 16; off > 0; off >>= 1) {
        s0 += __shfl_down(s0, off, 32);
        s1 += __shfl_down(s1, off, 32);
    }
    if (lane == 0) {
        EaQa[w]     += s0;                 // Ea
        EaQa[n + w] += s1;                 // Qa
        float o20 = s0 * s0, o21 = s1 * s1;
        cur2[w * 2 + 0] = o20;
        cur2[w * 2 + 1] = o21;
        if (!first) {
            float term = o20 / (o20 + last2[w * 2 + 0] + 1e-7f)
                       + o21 / (o21 + last2[w * 2 + 1] + 1e-7f);
            atomicAdd(nh, term / (2.0f * (float)n));
        }
    }
}

// ---------------------------------------------------------------------------
// Host orchestration.
// Input leaf order (setup_inputs dict / list insertion order):
//   0 Ra, 1 Za, 2 idx_i, 3 idx_j, 4 embedding_table, then 29 leaves per block:
//   0 Wg | 1,2 dense_i W,b | 3,4 dense_j W,b | 5..12 res_int (2x d1W,d1b,d2W,d2b)
//   13 u | 14,15 dense_o W,b | 16..23 res_atomic | 24..27 res_out | 28 Wout
// ---------------------------------------------------------------------------
extern "C" void kernel_launch(void* const* d_in, const int* in_sizes, int n_in,
                              void* d_out, int out_size, void* d_ws, size_t ws_size,
                              hipStream_t stream) {
    const float* Ra    = (const float*)d_in[0];
    const int*   Za    = (const int*)d_in[1];
    const int*   idx_i = (const int*)d_in[2];
    const int*   idx_j = (const int*)d_in[3];
    const float* emb   = (const float*)d_in[4];
    const int N = in_sizes[1];
    const int E = in_sizes[2];

    auto P = [&](int blk, int off) { return (const float*)d_in[5 + 29 * blk + off]; };

    size_t nf = (size_t)N * F_DIM;
    float* X   = (float*)d_ws;
    float* XI  = X   + nf;
    float* XJ  = XI  + nf;
    float* MSG = XJ  + nf;
    float* M   = MSG + nf;
    float* T   = M   + nf;
    float* H   = T   + nf;
    float* O2A = H   + nf;
    float* O2B = O2A + (size_t)N * 2;

    float* Ea = (float*)d_out;           // [N] Ea, [N] Qa, [1] nhloss
    float* nh = Ea + 2 * (size_t)N;

    (void)hipMemsetAsync(d_out, 0, (size_t)out_size * sizeof(float), stream);

    const int thr    = 256;
    const int g_elem = (int)((nf + thr - 1) / thr);
    const int g_add4 = (int)((nf / 4 + thr - 1) / thr);
    const int g_gemm = (N + 255) / 256;
    const int g_msg  = 2048;
    const int g_out  = (N * 32 + thr - 1) / thr;

    embed_kernel<<<g_elem, thr, 0, stream>>>(emb, Za, X, N);

    for (int b = 0; b < 3; b++) {
        const float *Wg = P(b,0);
        const float *Wi = P(b,1), *bi = P(b,2);
        const float *Wj = P(b,3), *bj = P(b,4);
        const float *u  = P(b,13);
        const float *Wo = P(b,14), *bo = P(b,15);
        const float *Wout = P(b,28);

        // interaction: xi = act(act(x)@Wi+bi), xj likewise
        gemm128_wmma<1,1,0><<<g_gemm, thr, 0, stream>>>(X, Wi, bi, nullptr, nullptr, XI, N);
        gemm128_wmma<1,1,0><<<g_gemm, thr, 0, stream>>>(X, Wj, bj, nullptr, nullptr, XJ, N);

        // messages: msg = segment_sum(g * xj[idx_j], idx_i)
        (void)hipMemsetAsync(MSG, 0, nf * sizeof(float), stream);
        msg_kernel<<<g_msg, thr, 0, stream>>>(Ra, idx_i, idx_j, Wg, XJ, MSG, E);

        // m = xi + msg
        add4_kernel<<<g_add4, thr, 0, stream>>>((const float4*)XI, (const float4*)MSG,
                                                (float4*)M, (int)(nf / 4));

        // res_int x2 (pre-activation residuals on M)
        for (int r = 0; r < 2; r++) {
            int o = 5 + 4 * r;
            gemm128_wmma<1,1,0><<<g_gemm, thr, 0, stream>>>(M, P(b,o+0), P(b,o+1), nullptr, nullptr, T, N);
            gemm128_wmma<0,0,1><<<g_gemm, thr, 0, stream>>>(T, P(b,o+2), P(b,o+3), M,       nullptr, M, N);
        }
        // x = u*x + act(m) @ Wo + bo
        gemm128_wmma<1,0,2><<<g_gemm, thr, 0, stream>>>(M, Wo, bo, X, u, X, N);

        // res_atomic x2 on X
        for (int r = 0; r < 2; r++) {
            int o = 16 + 4 * r;
            gemm128_wmma<1,1,0><<<g_gemm, thr, 0, stream>>>(X, P(b,o+0), P(b,o+1), nullptr, nullptr, T, N);
            gemm128_wmma<0,0,1><<<g_gemm, thr, 0, stream>>>(T, P(b,o+2), P(b,o+3), X,       nullptr, X, N);
        }
        // res_out x1: X -> H (X must survive for next block)
        gemm128_wmma<1,1,0><<<g_gemm, thr, 0, stream>>>(X, P(b,24), P(b,25), nullptr, nullptr, T, N);
        gemm128_wmma<0,0,1><<<g_gemm, thr, 0, stream>>>(T, P(b,26), P(b,27), X,       nullptr, H, N);

        // output head + nhloss ping-pong: b0->O2A, b1 reads O2A writes O2B, b2 reads O2B writes O2A
        float* curbuf = (b == 1) ? O2B : O2A;
        const float* lastbuf = (b == 1) ? O2A : O2B;
        out_kernel<<<g_out, thr, 0, stream>>>(H, Wout, Ea, nh, lastbuf, curbuf, N, b == 0 ? 1 : 0);
    }
}
